// FastSpatialWindowAttention_11012296147312
// MI455X (gfx1250) — compile-verified, matrix operand order checked by reading, not run on hardware
//
#include <hip/hip_runtime.h>
#include <hip/hip_bf16.h>

// CDNA5 / gfx1250 fused window-attention (Swin-style), bf16 WMMA pipeline.
// One workgroup per 8x8 window (4096 windows), 256 threads = 8 wave32's.
// Window input & output are moved by the Tensor Data Mover (TENSOR_LOAD_TO_LDS
// / TENSOR_STORE_FROM_LDS, TENSORcnt) with a 2D descriptor: 8 lines x 6KB.

typedef __attribute__((ext_vector_type(16))) __bf16 v16bf;
typedef __attribute__((ext_vector_type(8)))  __bf16 v8bf;
typedef __attribute__((ext_vector_type(4)))  __bf16 v4bf;
typedef __attribute__((ext_vector_type(8)))  float  v8f;
typedef __attribute__((ext_vector_type(4)))  unsigned int u32x4;
typedef __attribute__((ext_vector_type(8)))  int i32x8;

#define NHEADS 6
#define HDIM   32
#define CDIM   192
#define NTOK   64     // 8x8 tokens per window
#define NWIN   4096   // 4 * 32 * 32

// ---- fragment loader -------------------------------------------------------
// 16-bit A/B WMMA fragment (16x32 / 32x16): per the ISA layout, lane l holds
// row/col (l&15); elements 0..7 cover K = kbase + hi*8 .. +7 and elements
// 8..15 cover K = kbase+16+hi*8 .. +7  (hi = l>=16).  Both runs are 8
// contiguous bf16 = one 16-byte load each.
__device__ __forceinline__ v16bf frag16(const __bf16* rowptr, int kbase, int hi) {
    union { v16bf v; v8bf h[2]; } u;
    u.h[0] = *(const v8bf*)(rowptr + kbase + hi * 8);
    u.h[1] = *(const v8bf*)(rowptr + kbase + 16 + hi * 8);
    return u.v;
}

__device__ __forceinline__ v8f wmma_bf16(v16bf a, v16bf b, v8f c) {
    // D(f32 16x16) = A(bf16 16x32) * B(bf16 32x16) + C
    return __builtin_amdgcn_wmma_f32_16x16x32_bf16(
        /*neg_a=*/false, a, /*neg_b=*/false, b,
        /*c_mod=*/(short)0, c, /*reuse_a=*/false, /*reuse_b=*/false);
}

// ---- Tensor DMA descriptors (D#) -------------------------------------------
// 2D window tile: 8 lines of 1536 contiguous f32 (one 8-token image row x
// 192 ch), line stride = 256*192 f32.  tensor == tile (no OOB clipping).
__device__ __forceinline__ i32x8 tdm_group1_window() {
    i32x8 g1;
    g1[0] = 0x00020000;            // data_size=2 (4B); no multicast/pad/iterate
    g1[1] = (int)(1536u << 16);    // tensor_dim0[15:0] in [31:16]
    g1[2] = (int)(8u << 16);       // tensor_dim0[31:16]=0 | tensor_dim1[15:0]=8
    g1[3] = (int)(1536u << 16);    // tensor_dim1[31:16]=0 | tile_dim0=1536
    g1[4] = 8;                     // tile_dim1=8, tile_dim2=0
    g1[5] = 49152;                 // tensor_dim0_stride[31:0] = 256*192
    g1[6] = 0;                     // stride hi / dim1_stride lo
    g1[7] = 0;
    return g1;
}

__device__ __forceinline__ u32x4 tdm_group0(unsigned lds_addr, const void* gptr) {
    const unsigned long long ga = (unsigned long long)gptr;
    u32x4 g0;
    g0[0] = 1u;                                                  // count=1
    g0[1] = lds_addr;                                            // LDS byte addr
    g0[2] = (unsigned)ga;                                        // global[31:0]
    g0[3] = (unsigned)((ga >> 32) & 0x01FFFFFFull) | 0x80000000u; // [56:32]|type=2
    return g0;
}

__device__ __forceinline__ void tdm_load(unsigned lds_addr, const void* gsrc) {
    const u32x4 g0 = tdm_group0(lds_addr, gsrc);
    const i32x8 g1 = tdm_group1_window();
    asm volatile("tensor_load_to_lds %0, %1" :: "s"(g0), "s"(g1) : "memory");
}

__device__ __forceinline__ void tdm_store(unsigned lds_addr, void* gdst) {
    const u32x4 g0 = tdm_group0(lds_addr, gdst);
    const i32x8 g1 = tdm_group1_window();
    asm volatile("tensor_store_from_lds %0, %1" :: "s"(g0), "s"(g1) : "memory");
}

// ---- weight precast kernel -------------------------------------------------
__global__ void precast_weights(const float* __restrict__ qkv_w,
                                const float* __restrict__ proj_w,
                                __bf16* __restrict__ wqkv,
                                __bf16* __restrict__ wproj) {
    int i = blockIdx.x * 256 + threadIdx.x;
    if (i < 576 * 192) wqkv[i]  = (__bf16)qkv_w[i];
    if (i < 192 * 192) wproj[i] = (__bf16)proj_w[i];
}

// ---- fused window attention ------------------------------------------------
// Dynamic LDS layout (bf16 element offsets):
//   stage[64][192] f32 @ 0      (49152 B)  TDM in-stage  (dead after ph.1)
//   sQ   [6][64][32]  @ 0       Q (pre-scaled 1/sqrt(32))   \
//   sK   [6][64][32]  @ 12288                                } written ph.2
//   sVt  [6][32][64]  @ 24576   V transposed (d-major)      /
//   ostage[64][192]f32 @ 0      (49152 B)  TDM out-stage (reuses sQ/K, ph.4)
//   sX   [64][192]    @ 36864   window input bf16 (dead after QKV GEMM)
//   sPw  [8][16][64]  @ 36864   per-wave softmax P slab (reuses sX)
//   sO   [64][192]    @ 45056   attention output (reuses sX tail)
// total 57344 bf16 = 114688 bytes
__global__ void __launch_bounds__(256, 2)
window_attn_kernel(const float*  __restrict__ x,
                   const __bf16* __restrict__ wqkv,
                   const float*  __restrict__ qkv_b,
                   const __bf16* __restrict__ wproj,
                   const float*  __restrict__ proj_b,
                   const float*  __restrict__ bias,
                   float*        __restrict__ out) {
    extern __shared__ __bf16 smem[];
    float*  stage  = (float*)smem;       // phase-1 in-stage (overlaps sQ/sK)
    float*  ostage = (float*)smem;       // phase-4 out-stage (overlaps sQ/sK)
    __bf16* sQ  = smem;
    __bf16* sK  = smem + 12288;
    __bf16* sVt = smem + 24576;
    __bf16* sX  = smem + 36864;
    __bf16* sPw = smem + 36864;
    __bf16* sO  = smem + 45056;

    const int tid  = threadIdx.x;
    const int lane = tid & 31;
    const int wave = tid >> 5;
    const int lrow = lane & 15;   // M (A) or N (B/C) within a 16-wide tile
    const int hi   = lane >> 4;   // K-half selector for A/B fragments

    const int w   = blockIdx.x;        // window id
    const int b   = w >> 10;
    const int rem = w & 1023;
    const int wh  = rem >> 5;
    const int ww  = rem & 31;
    const long winbase = (long)b * 65536 + (long)(wh * 8) * 256 + (long)(ww * 8);

    // ---- Phase 1a: TDM DMA window x (f32, 8 lines x 6KB) -> LDS stage ------
    if (wave == 0) {
        tdm_load((unsigned)(unsigned long long)(void*)stage, x + winbase * CDIM);
        __builtin_amdgcn_s_wait_tensorcnt(0);
    }
    __syncthreads();

    // ---- Phase 1b: convert stage f32 -> bf16 sX (packed b64 stores) --------
    for (int i = tid; i < NTOK * (CDIM / 4); i += 256) {
        const float4 v = *(const float4*)(stage + i * 4);
        v4bf o;
        o[0] = (__bf16)v.x; o[1] = (__bf16)v.y;
        o[2] = (__bf16)v.z; o[3] = (__bf16)v.w;
        *(v4bf*)(sX + i * 4) = o;
    }
    __syncthreads();

    // ---- Phase 2: QKV GEMM  [64,192] x [192,576]^T -------------------------
    // 4 m-tiles x 36 n-tiles, 6 K-steps each; waves round-robin over tiles.
    for (int tile = wave; tile < 144; tile += 8) {
        const int mt = tile & 3;
        const int nt = tile >> 2;            // 0..35
        const __bf16* arow = sX + (mt * 16 + lrow) * CDIM;
        const __bf16* brow = wqkv + (nt * 16 + lrow) * CDIM;
        __builtin_prefetch(brow, 0, 3);      // -> global_prefetch_b8 (L2-resident weights)
        v8f acc = {};
        #pragma unroll
        for (int kb = 0; kb < 6; ++kb) {
            v16bf A = frag16(arow, kb * 32, hi);
            v16bf B = frag16(brow, kb * 32, hi);
            acc = wmma_bf16(A, B, acc);
        }
        // epilogue: +bias, scale Q, branch-free scatter into Q / K / V^T
        const int which = nt / 12;             // 0=q 1=k 2=v
        const int nt12  = nt % 12;
        const int head  = nt12 >> 1;
        const int d     = (nt12 & 1) * 16 + lrow;
        const float bn  = qkv_b[nt * 16 + lrow];
        const float mul = (which == 0) ? 0.17677669529663687f : 1.0f; // 32^-0.5
        const int t0    = mt * 16 + ((lane < 16) ? 0 : 8);
        const bool isV  = (which == 2);
        __bf16* dstQK = smem + which * 12288 + (head * 64 + t0) * 32 + d;
        __bf16* dstV  = sVt + (head * 32 + d) * 64 + t0;
        __bf16* dst   = isV ? dstV : dstQK;
        const int stride = isV ? 1 : 32;
        #pragma unroll
        for (int j = 0; j < 8; ++j)
            dst[j * stride] = (__bf16)((acc[j] + bn) * mul);
    }
    __syncthreads();

    // ---- Phase 3: per-head attention (waves 0..5, one head each) -----------
    if (wave < NHEADS) {
        const int head = wave;
        const __bf16* Qh = sQ  + head * 64 * 32;
        const __bf16* Kh = sK  + head * 64 * 32;
        const __bf16* Vh = sVt + head * 32 * 64;
        const float*  Bh = bias + head * 64 * 64;
        __bf16* P = sPw + wave * 16 * 64;

        for (int qt = 0; qt < 4; ++qt) {
            // S tile row: q-tile x 4 k-tiles, K = head_dim = 32 -> 1 WMMA each
            const v16bf Aq = frag16(Qh + (qt * 16 + lrow) * 32, 0, hi);
            v8f s[4];
            #pragma unroll
            for (int kt = 0; kt < 4; ++kt) {
                v16bf Bk = frag16(Kh + (kt * 16 + lrow) * 32, 0, hi);
                v8f z = {};
                s[kt] = wmma_bf16(Aq, Bk, z);
            }
            // softmax over 64 keys per query row (+ relative position bias).
            // C layout: vgpr j -> row M=j (lanes 0-15) / M=j+8 (lanes 16-31),
            // key = kt*16 + (lane&15).  xor-shuffles <16 stay in each half.
            const int Mbase = (lane < 16) ? 0 : 8;
            #pragma unroll
            for (int j = 0; j < 8; ++j) {
                const int q = qt * 16 + Mbase + j;
                float v0 = s[0][j] + Bh[q * 64 +  0 + lrow];
                float v1 = s[1][j] + Bh[q * 64 + 16 + lrow];
                float v2 = s[2][j] + Bh[q * 64 + 32 + lrow];
                float v3 = s[3][j] + Bh[q * 64 + 48 + lrow];
                float mx = fmaxf(fmaxf(v0, v1), fmaxf(v2, v3));
                #pragma unroll
                for (int m = 1; m < 16; m <<= 1)
                    mx = fmaxf(mx, __shfl_xor(mx, m, 32));
                const float e0 = __expf(v0 - mx), e1 = __expf(v1 - mx);
                const float e2 = __expf(v2 - mx), e3 = __expf(v3 - mx);
                float sum = (e0 + e1) + (e2 + e3);
                #pragma unroll
                for (int m = 1; m < 16; m <<= 1)
                    sum += __shfl_xor(sum, m, 32);
                const float inv = __frcp_rn(sum);
                __bf16* prow = P + (Mbase + j) * 64 + lrow;
                prow[ 0] = (__bf16)(e0 * inv);
                prow[16] = (__bf16)(e1 * inv);
                prow[32] = (__bf16)(e2 * inv);
                prow[48] = (__bf16)(e3 * inv);
            }
            // wave-local LDS RAW: P written above, read as A-fragments below
            asm volatile("s_wait_dscnt 0" ::: "memory");

            // O tile = P[16,64] x V[64,32]  (2 n-tiles x 2 K-steps)
            const __bf16* prowA = P + lrow * 64;
            #pragma unroll
            for (int ntd = 0; ntd < 2; ++ntd) {
                v8f o = {};
                #pragma unroll
                for (int kb = 0; kb < 2; ++kb) {
                    v16bf Ap = frag16(prowA, kb * 32, hi);
                    v16bf Bv = frag16(Vh + (ntd * 16 + lrow) * 64, kb * 32, hi);
                    o = wmma_bf16(Ap, Bv, o);
                }
                __bf16* obase = sO + (qt * 16 + Mbase) * CDIM + head * 32 + ntd * 16 + lrow;
                #pragma unroll
                for (int j = 0; j < 8; ++j)
                    obase[j * CDIM] = (__bf16)o[j];
            }
        }
    }
    __syncthreads();

    // ---- Phase 4: output projection -> LDS out-stage (f32) -----------------
    for (int tile = wave; tile < 48; tile += 8) {
        const int mt = tile & 3;
        const int nt = tile >> 2;            // 0..11
        const __bf16* arow = sO + (mt * 16 + lrow) * CDIM;
        const __bf16* brow = wproj + (nt * 16 + lrow) * CDIM;
        __builtin_prefetch(brow, 0, 3);
        v8f acc = {};
        #pragma unroll
        for (int kb = 0; kb < 6; ++kb) {
            v16bf A = frag16(arow, kb * 32, hi);
            v16bf B = frag16(brow, kb * 32, hi);
            acc = wmma_bf16(A, B, acc);
        }
        const float pb = proj_b[nt * 16 + lrow];
        const int t0 = mt * 16 + ((lane < 16) ? 0 : 8);
        float* obase = ostage + t0 * CDIM + nt * 16 + lrow;
        #pragma unroll
        for (int j = 0; j < 8; ++j)
            obase[j * CDIM] = acc[j] + pb;
    }
    __syncthreads();

    // ---- Phase 5: TDM DMA out-stage -> global (8 lines x 6KB) --------------
    if (wave == 0) {
        tdm_store((unsigned)(unsigned long long)(void*)ostage, out + winbase * CDIM);
        __builtin_amdgcn_s_wait_tensorcnt(0);  // S_ENDPGM also implies wait-idle
    }
}

// ---- host launch -----------------------------------------------------------
extern "C" void kernel_launch(void* const* d_in, const int* in_sizes, int n_in,
                              void* d_out, int out_size, void* d_ws, size_t ws_size,
                              hipStream_t stream) {
    const float* x      = (const float*)d_in[0];
    const float* qkv_w  = (const float*)d_in[1];
    const float* qkv_b  = (const float*)d_in[2];
    const float* proj_w = (const float*)d_in[3];
    const float* proj_b = (const float*)d_in[4];
    const float* bias   = (const float*)d_in[5];
    float* out = (float*)d_out;

    __bf16* wqkv  = (__bf16*)d_ws;                       // 576*192 bf16
    __bf16* wproj = wqkv + 576 * 192;                    // 192*192 bf16

    precast_weights<<<(576 * 192 + 255) / 256, 256, 0, stream>>>(qkv_w, proj_w, wqkv, wproj);

    const size_t smem_bytes = 57344 * sizeof(__bf16);    // 114688 B
    window_attn_kernel<<<NWIN, 256, smem_bytes, stream>>>(
        x, wqkv, qkv_b, wproj, proj_b, bias, out);

    (void)in_sizes; (void)n_in; (void)out_size; (void)ws_size;
}
